// MultiHeadAttention_88665304858999
// MI455X (gfx1250) — matrix operand order repaired and checked by reading, not compiled, not run on hardware
//
#include <hip/hip_runtime.h>

typedef __attribute__((ext_vector_type(16))) _Float16 v16h;
typedef __attribute__((ext_vector_type(8)))  _Float16 v8h;
typedef __attribute__((ext_vector_type(8)))  float    v8f;
typedef __attribute__((ext_vector_type(4)))  int      v4i;

#define B_  16
#define N_  1024
#define E_  768
#define H_  8
#define D_  96
#define E3_ 2304

#if defined(__HIP_DEVICE_COMPILE__) && __has_builtin(__builtin_amdgcn_global_load_async_to_lds_b128)
#define HAVE_ASYNC_LDS 1
#else
#define HAVE_ASYNC_LDS 0
#endif

__device__ __forceinline__ v8f wmma16(v16h a, v16h b, v8f c) {
  // D = A(16x32 f16) * B(32x16 f16) + C(16x16 f32)
  return __builtin_amdgcn_wmma_f32_16x16x32_f16(false, a, false, b, (short)0, c,
                                                false, false);
}

__device__ __forceinline__ v16h pack16(v8h lo, v8h hi) {
  v16h r;
#pragma unroll
  for (int j = 0; j < 8; ++j) { r[j] = lo[j]; r[j + 8] = hi[j]; }
  return r;
}

__device__ __forceinline__ v8f zero8() {
  v8f z = {0.f, 0.f, 0.f, 0.f, 0.f, 0.f, 0.f, 0.f};
  return z;
}

// A-operand (16x32 f16) per ISA: lane<16 row m=lane holds K 0..7,16..23;
// lane>=16 row m=lane-16 holds K 8..15,24..31.
__device__ __forceinline__ v16h load_a_f16(const _Float16* rowp, int kb, int hi) {
  v8h lo = *(const v8h*)(rowp + kb + 8 * hi);
  v8h hh = *(const v8h*)(rowp + kb + 16 + 8 * hi);
  return pack16(lo, hh);
}

// B-operand (32x16 f16): lane L holds column n=L&15, K 8*(L>>4)+0..7 and 16+8*(L>>4)+0..7 (same order as A).
__device__ __forceinline__ v16h load_b_f16(const _Float16* rowp, int kb, int hi) {
  v8h lo = *(const v8h*)(rowp + kb + 8 * hi);
  v8h hh = *(const v8h*)(rowp + kb + 16 + 8 * hi);
  return pack16(lo, hh);
}

// 16-byte global -> LDS copy; async (ASYNCcnt-tracked, no VGPR round trip) when available.
__device__ __forceinline__ void copy16_g2l(_Float16* lds_dst, const _Float16* gsrc) {
#if HAVE_ASYNC_LDS
  typedef __attribute__((address_space(1))) v4i gv4i;
  typedef __attribute__((address_space(3))) v4i lv4i;
  __builtin_amdgcn_global_load_async_to_lds_b128(
      (gv4i*)(uintptr_t)gsrc,
      (lv4i*)(unsigned int)(uintptr_t)lds_dst,
      0, 0);
#else
  *(v8h*)lds_dst = *(const v8h*)gsrc;
#endif
}

__device__ __forceinline__ void async_join() {
#if HAVE_ASYNC_LDS
  asm volatile("s_wait_asynccnt 0x0" ::: "memory");
#endif
}

// ---------------- Kernel 0: f32 -> f16 conversion (x, qkv_w, proj_w) ----------------
__global__ __launch_bounds__(256) void cvt_kernel(const float* __restrict__ src,
                                                  _Float16* __restrict__ dst, int n8) {
  int i = blockIdx.x * 256 + threadIdx.x;
  if (i >= n8) return;
  const float4* p = (const float4*)src + (size_t)i * 2;
  float4 a = p[0], b = p[1];
  v8h o;
  o[0] = (_Float16)a.x; o[1] = (_Float16)a.y; o[2] = (_Float16)a.z; o[3] = (_Float16)a.w;
  o[4] = (_Float16)b.x; o[5] = (_Float16)b.y; o[6] = (_Float16)b.z; o[7] = (_Float16)b.w;
  *(v8h*)(dst + (size_t)i * 8) = o;
}

// ---------------- Kernel 1: QKV projection + scatter ----------------
// C[m, c] = sum_k x[m,k] * qkv_w[c,k] + qkv_b[c], scatter by c=(h*96+d)*3+comp.
__global__ __launch_bounds__(256) void qkv_kernel(
    const _Float16* __restrict__ xh, const _Float16* __restrict__ wh,
    const float* __restrict__ bias, _Float16* __restrict__ Q,
    _Float16* __restrict__ K, _Float16* __restrict__ V) {
  __shared__ __align__(16) _Float16 Xs[16][E_];
  const int tid = threadIdx.x;
  const int rowbase = blockIdx.x * 16;

  const _Float16* xrow = xh + (size_t)rowbase * E_;
  for (int i = tid; i < (16 * E_) / 8; i += 256)        // 1536 x 16B, verbatim
    copy16_g2l(&Xs[0][0] + i * 8, xrow + (size_t)i * 8);
  async_join();
  __syncthreads();

  const int wid = tid >> 5, lane = tid & 31;
  const int l15 = lane & 15, hi = lane >> 4;
  const int col = blockIdx.y * 128 + wid * 16 + l15;    // this lane's output column (B op)

  v8f acc = zero8();
  const _Float16* wrow = wh + (size_t)col * E_;
#pragma unroll 4
  for (int kb = 0; kb < E_; kb += 32) {
    v16h a = load_a_f16(&Xs[l15][0], kb, hi);
    v16h b = load_b_f16(wrow, kb, hi);
    acc = wmma16(a, b, acc);
  }

  const float bb = bias[col];
  const int comp = col % 3;
  const int hd = col / 3;
  const int h = hd / D_, d = hd - h * D_;
  _Float16* dst = (comp == 0) ? Q : ((comp == 1) ? K : V);
#pragma unroll
  for (int r = 0; r < 8; ++r) {
    int row = rowbase + r + 8 * hi;        // global token index in [0, B*N)
    int bi = row >> 10, nseq = row & 1023;
    dst[(((size_t)(bi * H_ + h)) * N_ + nseq) * D_ + d] = (_Float16)(acc[r] + bb);
  }
}

// ---------------- Kernel 2: flash attention ----------------
// block = (qtile, h, b); 8 waves = 4 q-groups x 2 D-halves (48 cols each)
__global__ __launch_bounds__(256) void attn_kernel(
    const _Float16* __restrict__ Q, const _Float16* __restrict__ K,
    const _Float16* __restrict__ V, _Float16* __restrict__ Aout) {
  __shared__ __align__(16) _Float16 Ks[32][D_];
  __shared__ __align__(16) _Float16 Vt[D_][32];   // transposed V tile
  __shared__ __align__(16) _Float16 Ps[8][16][32];

  const int tid = threadIdx.x;
  const int wid = tid >> 5, lane = tid & 31;
  const int l15 = lane & 15, hi = lane >> 4;
  const int qg = wid >> 1, dh = wid & 1;
  const int dbase = dh * 48;
  const int b = blockIdx.z, h = blockIdx.y;
  const size_t headoff = ((size_t)(b * H_ + h)) * N_ * D_;
  const int qrow0 = blockIdx.x * 64 + qg * 16;

  // Preload Q fragments (16x96 per q-group), reused for all key tiles.
  v16h qa[3];
#pragma unroll
  for (int s = 0; s < 3; ++s)
    qa[s] = load_a_f16(Q + headoff + (size_t)(qrow0 + l15) * D_, s * 32, hi);

  float mrow[8], lrow[8];
  v8f o[3];
#pragma unroll
  for (int r = 0; r < 8; ++r) { mrow[r] = -1e30f; lrow[r] = 0.f; }
#pragma unroll
  for (int c = 0; c < 3; ++c) o[c] = zero8();

  for (int kt = 0; kt < N_; kt += 32) {
    __syncthreads();
    // K tile: verbatim row-major 32x96 copy -> async path
    const _Float16* kbase = K + headoff + (size_t)kt * D_;
    for (int i = tid; i < (32 * D_) / 8; i += 256)       // 384 x 16B
      copy16_g2l(&Ks[0][0] + i * 8, kbase + (size_t)i * 8);
    // V tile transposed (cannot async-copy a transpose)
    const _Float16* vbase = V + headoff + (size_t)kt * D_;
    for (int i = tid; i < 32 * D_; i += 256) {
      int l = i / D_, d = i - l * D_;
      Vt[d][l] = vbase[(size_t)l * D_ + d];
    }
    async_join();
    __syncthreads();

    // S = Q * K^T (16 x 32), two 16-col subtiles, K-dim = D = 3*32
    v8f s0 = zero8(), s1 = zero8();
#pragma unroll
    for (int s = 0; s < 3; ++s) {
      int db = s * 32;
      v16h b0 = load_b_f16(&Ks[l15][0], db, hi);
      v16h b1 = load_b_f16(&Ks[16 + l15][0], db, hi);
      s0 = wmma16(qa[s], b0, s0);
      s1 = wmma16(qa[s], b1, s1);
    }

    // Online softmax (no pre-scale; reference scales AFTER softmax).
#pragma unroll
    for (int r = 0; r < 8; ++r) {
      float mx = fmaxf(s0[r], s1[r]);
#pragma unroll
      for (int off = 8; off >= 1; off >>= 1)
        mx = fmaxf(mx, __shfl_xor(mx, off, 16));
      float mnew = fmaxf(mrow[r], mx);
      float corr = __expf(mrow[r] - mnew);
      float p0 = __expf(s0[r] - mnew);
      float p1 = __expf(s1[r] - mnew);
      s0[r] = p0; s1[r] = p1;
      float rs = p0 + p1;
#pragma unroll
      for (int off = 8; off >= 1; off >>= 1)
        rs += __shfl_xor(rs, off, 16);
      lrow[r] = lrow[r] * corr + rs;
      mrow[r] = mnew;
      o[0][r] *= corr; o[1][r] *= corr; o[2][r] *= corr;
    }

    // P through LDS to acquire the A-operand swizzle (per-wave region;
    // LDS ops from the same wave are processed in order).
#pragma unroll
    for (int r = 0; r < 8; ++r) {
      int m = r + 8 * hi;
      Ps[wid][m][l15]      = (_Float16)s0[r];
      Ps[wid][m][16 + l15] = (_Float16)s1[r];
    }
    __builtin_amdgcn_wave_barrier();

    v16h pa = load_a_f16(&Ps[wid][l15][0], 0, hi);
#pragma unroll
    for (int c = 0; c < 3; ++c) {
      int dcol = dbase + c * 16 + l15;
      v16h vb = load_b_f16(&Vt[dcol][0], 0, hi);  // K over l, contiguous in Vt row
      o[c] = wmma16(pa, vb, o[c]);
    }
  }

  // out = (softmax / sqrt(E)) @ V  ->  O * 1/(l * sqrt(768))
  const float scl = 0.03608439182435161f;  // 1/sqrt(768)
#pragma unroll
  for (int r = 0; r < 8; ++r) {
    float inv = scl / lrow[r];
    int qrow = qrow0 + r + 8 * hi;
#pragma unroll
    for (int c = 0; c < 3; ++c) {
      int d = dbase + c * 16 + l15;
      Aout[((size_t)(b * N_ + qrow)) * E_ + h * D_ + d] = (_Float16)(o[c][r] * inv);
    }
  }
}

// ---------------- Kernel 3: output projection ----------------
__global__ __launch_bounds__(256) void proj_kernel(
    const _Float16* __restrict__ A, const _Float16* __restrict__ wh,
    const float* __restrict__ bias, float* __restrict__ out) {
  __shared__ __align__(16) _Float16 Xs[16][E_];
  const int tid = threadIdx.x;
  const int rowbase = blockIdx.x * 16;

  const _Float16* arow = A + (size_t)rowbase * E_;
  for (int i = tid; i < (16 * E_) / 8; i += 256)
    copy16_g2l(&Xs[0][0] + i * 8, arow + (size_t)i * 8);
  async_join();
  __syncthreads();

  const int wid = tid >> 5, lane = tid & 31;
  const int l15 = lane & 15, hi = lane >> 4;
  const int col = blockIdx.y * 128 + wid * 16 + l15;

  v8f acc = zero8();
  const _Float16* wrow = wh + (size_t)col * E_;
#pragma unroll 4
  for (int kb = 0; kb < E_; kb += 32) {
    v16h a = load_a_f16(&Xs[l15][0], kb, hi);
    v16h b = load_b_f16(wrow, kb, hi);
    acc = wmma16(a, b, acc);
  }

  const float bb = bias[col];
#pragma unroll
  for (int r = 0; r < 8; ++r) {
    int row = rowbase + r + 8 * hi;
    out[(size_t)row * E_ + col] = acc[r] + bb;
  }
}

extern "C" void kernel_launch(void* const* d_in, const int* in_sizes, int n_in,
                              void* d_out, int out_size, void* d_ws, size_t ws_size,
                              hipStream_t stream) {
  (void)in_sizes; (void)n_in; (void)out_size; (void)ws_size;
  const float* x      = (const float*)d_in[0];
  const float* qkv_w  = (const float*)d_in[1];
  const float* qkv_b  = (const float*)d_in[2];
  const float* proj_w = (const float*)d_in[3];
  const float* proj_b = (const float*)d_in[4];
  float* out = (float*)d_out;

  const size_t he = (size_t)B_ * H_ * N_ * D_;   // 12,582,912 elements
  _Float16* Q   = (_Float16*)d_ws;
  _Float16* Kp  = Q + he;
  _Float16* Vp  = Kp + he;
  _Float16* At  = Vp + he;
  _Float16* Xh  = At + he;
  _Float16* Wqh = Xh + he;                       // 2304*768
  _Float16* Wph = Wqh + (size_t)E3_ * E_;        // 768*768

  const int nx = B_ * N_ * E_ / 8;               // 1,572,864
  const int nq = E3_ * E_ / 8;                   // 221,184
  const int np = E_ * E_ / 8;                    // 73,728
  cvt_kernel<<<dim3((nx + 255) / 256), 256, 0, stream>>>(x, Xh, nx);
  cvt_kernel<<<dim3((nq + 255) / 256), 256, 0, stream>>>(qkv_w, Wqh, nq);
  cvt_kernel<<<dim3((np + 255) / 256), 256, 0, stream>>>(proj_w, Wph, np);

  qkv_kernel <<<dim3(B_ * N_ / 16, E3_ / 128), 256, 0, stream>>>(Xh, Wqh, qkv_b, Q, Kp, Vp);
  attn_kernel<<<dim3(N_ / 64, H_, B_),          256, 0, stream>>>(Q, Kp, Vp, At);
  proj_kernel<<<dim3(B_ * N_ / 16, E_ / 128),   256, 0, stream>>>(At, Wph, proj_b, out);
}
